// AttentionLayer_41549513621918
// MI455X (gfx1250) — matrix operand order repaired and checked by reading, not compiled, not run on hardware
//
#include <hip/hip_runtime.h>
#include <hip/hip_bf16.h>

// ---------------------------------------------------------------------------
// Problem constants (from the reference)
// ---------------------------------------------------------------------------
#define NB    4
#define NL    4096
#define NK    32
#define NDIN  128
#define NU    128
#define NLAT  64
#define NBL   (NB * NL)

typedef _Float16 hvec8  __attribute__((ext_vector_type(8)));
typedef _Float16 hvec16 __attribute__((ext_vector_type(16)));
typedef float    fvec8  __attribute__((ext_vector_type(8)));

// Build a 16-half WMMA fragment from two contiguous 16-byte chunks.
__device__ __forceinline__ hvec16 ld_frag(const _Float16* p_lo, const _Float16* p_hi) {
    hvec8 lo = *(const hvec8*)p_lo;
    hvec8 hi = *(const hvec8*)p_hi;
    return __builtin_shufflevector(lo, hi, 0, 1, 2, 3, 4, 5, 6, 7,
                                           8, 9, 10, 11, 12, 13, 14, 15);
}

// ---------------------------------------------------------------------------
// Kernel 0: convert MLP weights to transposed f16 (so B fragments are
// two contiguous 8-half runs per lane: lane = output column n, k contiguous).
//   W0T[n][k] = Wa0[k][n]   n<64,  k<128
//   W1T[n][k] = Wa1[k][n]   n<128, k<64
// ---------------------------------------------------------------------------
__global__ void prep_weights(const float* __restrict__ Wa0,
                             const float* __restrict__ Wa1,
                             _Float16* __restrict__ W0T,
                             _Float16* __restrict__ W1T) {
    int t = blockIdx.x * blockDim.x + threadIdx.x;   // 8192 threads
    if (t < NU * NLAT) {
        int n = t / NU, k = t % NU;                  // [64][128]
        W0T[n * NU + k] = (_Float16)Wa0[k * NLAT + n];
    }
    if (t < NLAT * NU) {
        int n = t / NLAT, k = t % NLAT;              // [128][64]
        W1T[n * NLAT + k] = (_Float16)Wa1[k * NU + n];
    }
}

// ---------------------------------------------------------------------------
// Kernel 1: q/k/v projections, fp32 (tiny GEMM; weights are L2 resident).
// One block per row, one thread per output channel.
// ---------------------------------------------------------------------------
__global__ __launch_bounds__(NU)
void proj_qkv(const float* __restrict__ x,
              const float* __restrict__ Wq,
              const float* __restrict__ Wk,
              const float* __restrict__ Wv,
              float* __restrict__ Q,
              float* __restrict__ Kp,
              float* __restrict__ Vp) {
    __shared__ float sx[NDIN];
    const size_t row = blockIdx.x;
    const int c = threadIdx.x;
    sx[c] = x[row * NDIN + c];
    __syncthreads();
    float aq = 0.f, ak = 0.f, av = 0.f;
#pragma unroll 8
    for (int d = 0; d < NDIN; ++d) {
        float xv = sx[d];
        aq += xv * Wq[d * NU + c];
        ak += xv * Wk[d * NU + c];
        av += xv * Wv[d * NU + c];
    }
    Q [row * NU + c] = aq;
    Kp[row * NU + c] = ak;
    Vp[row * NU + c] = av;
}

// ---------------------------------------------------------------------------
// Kernel 2: fused gather + attention MLP (WMMA f16) + per-channel softmax +
// aggregation. One 128-thread (4-wave) block per (b,l) site.
// ---------------------------------------------------------------------------
__global__ __launch_bounds__(128)
void attn_main(const float* __restrict__ pe,
               const int*   __restrict__ nn_idx,
               const float* __restrict__ Q,
               const float* __restrict__ Kp,
               const float* __restrict__ Vp,
               const _Float16* __restrict__ W0T,
               const _Float16* __restrict__ W1T,
               float* __restrict__ out) {
    constexpr int SS  = 132;  // f32 logits row stride (pad: banks differ between M halves)
    constexpr int SA0 = 136;  // f16 A0 row stride (272B rows -> 16B aligned)
    constexpr int SH  = 72;   // f16 hidden row stride (144B rows -> 16B aligned)

    __shared__ float sPE [NK * NU];          // 16 KB
    __shared__ float sVal[NK * NU];          // 16 KB
    __shared__ float sS  [NK * SS];          // 16.5 KB  (logits -> softmax weights)
    __shared__ __align__(16) _Float16 sA0[NK * SA0];  // 8.5 KB  (q - k + pe)
    __shared__ __align__(16) _Float16 sH [NK * SH];   // 4.5 KB  (relu hidden)
    __shared__ int sIdx[NK];

    const int tid = threadIdx.x;
    const int bl  = blockIdx.x;          // 0 .. B*L-1
    const int b   = bl >> 12;            // L = 4096

    // ---- Phase 1: gather neighbors, stage pe/val, build A0 = (q - k + pe) ----
    if (tid < NK) sIdx[tid] = nn_idx[(size_t)bl * NK + tid];
    __syncthreads();

    const float  qc  = Q[(size_t)bl * NU + tid];
    const float* peB = pe + (size_t)bl * NK * NU;
#pragma unroll 4
    for (int k = 0; k < NK; ++k) {
        const size_t nrow = (size_t)(b * NL + sIdx[k]);
        float pev = peB[k * NU + tid];               // HBM (dominant traffic)
        float kv  = Kp[nrow * NU + tid];             // L2 resident
        float vv  = Vp[nrow * NU + tid];             // L2 resident
        sPE [k * NU  + tid] = pev;
        sVal[k * NU  + tid] = vv;
        sA0 [k * SA0 + tid] = (_Float16)(qc - kv + pev);
    }
    __syncthreads();

    const int lane = tid & 31;
    const int wave = tid >> 5;
    const int half = lane >> 4;
    const int lm   = lane & 15;

    // ---- Phase 2: layer 0  H = relu(A0[32,128] @ W0[128,64]) via WMMA ----
    {
        const int ntile = wave;                      // 4 waves cover N=64
#pragma unroll
        for (int mtile = 0; mtile < 2; ++mtile) {
            fvec8 acc = {};
#pragma unroll
            for (int kt = 0; kt < 4; ++kt) {
                const int arow = mtile * 16 + lm;
                const int acol = kt * 32 + half * 8;      // lanes0-15: k0-7/16-23; lanes16-31: k8-15/24-31
                hvec16 a = ld_frag(&sA0[arow * SA0 + acol],
                                   &sA0[arow * SA0 + acol + 16]);
                const int brow = ntile * 16 + lm;         // output column
                const int bcol = kt * 32 + half * 8;      // lanes0-15: k0-7/16-23; lanes16-31: k8-15/24-31
                hvec16 bf = ld_frag(&W0T[brow * NU + bcol],
                                    &W0T[brow * NU + bcol + 16]);
                acc = __builtin_amdgcn_wmma_f32_16x16x32_f16(
                          false, a, false, bf, (short)0, acc, false, false);
            }
#pragma unroll
            for (int d = 0; d < 8; ++d) {
                const int m = mtile * 16 + half * 8 + d;  // C/D layout: vgpr d -> M = d (+8 for hi half)
                const int n = ntile * 16 + lm;
                sH[m * SH + n] = (_Float16)fmaxf(acc[d], 0.0f);
            }
        }
    }
    __syncthreads();

    // ---- Phase 3: layer 1  S = H[32,64] @ W1[64,128] via WMMA ----
#pragma unroll
    for (int nt2 = 0; nt2 < 2; ++nt2) {
        const int ntile = wave * 2 + nt2;            // 4 waves x 2 cover N=128
#pragma unroll
        for (int mtile = 0; mtile < 2; ++mtile) {
            fvec8 acc = {};
#pragma unroll
            for (int kt = 0; kt < 2; ++kt) {
                const int arow = mtile * 16 + lm;
                const int acol = kt * 32 + half * 8;
                hvec16 a = ld_frag(&sH[arow * SH + acol],
                                   &sH[arow * SH + acol + 16]);
                const int brow = ntile * 16 + lm;
                const int bcol = kt * 32 + half * 8;
                hvec16 bf = ld_frag(&W1T[brow * NLAT + bcol],
                                    &W1T[brow * NLAT + bcol + 16]);
                acc = __builtin_amdgcn_wmma_f32_16x16x32_f16(
                          false, a, false, bf, (short)0, acc, false, false);
            }
#pragma unroll
            for (int d = 0; d < 8; ++d) {
                const int m = mtile * 16 + half * 8 + d;
                const int n = ntile * 16 + lm;
                sS[m * SS + n] = acc[d];
            }
        }
    }
    __syncthreads();

    // ---- Phase 4: per-channel softmax over K + weighted aggregation ----
    {
        const int c = tid;                           // one channel per thread
        float mx = -3.0e38f;
#pragma unroll 8
        for (int k = 0; k < NK; ++k) mx = fmaxf(mx, sS[k * SS + c]);
        float sum = 0.f;
#pragma unroll 8
        for (int k = 0; k < NK; ++k) {
            float e = __expf(sS[k * SS + c] - mx);
            sS[k * SS + c] = e;
            sum += e;
        }
        const float inv = 1.0f / sum;
        float o = 0.f;
#pragma unroll 8
        for (int k = 0; k < NK; ++k)
            o += (sVal[k * NU + c] + sPE[k * NU + c]) * (sS[k * SS + c] * inv);
        out[(size_t)bl * NU + c] = o;
    }
}

// ---------------------------------------------------------------------------
// Launch
// ---------------------------------------------------------------------------
extern "C" void kernel_launch(void* const* d_in, const int* in_sizes, int n_in,
                              void* d_out, int out_size, void* d_ws, size_t ws_size,
                              hipStream_t stream) {
    const float* x   = (const float*)d_in[0];
    const float* pe  = (const float*)d_in[1];
    const int*   nn  = (const int*)  d_in[2];
    const float* Wq  = (const float*)d_in[3];
    const float* Wk  = (const float*)d_in[4];
    const float* Wv  = (const float*)d_in[5];
    const float* Wa0 = (const float*)d_in[6];
    const float* Wa1 = (const float*)d_in[7];
    float* out = (float*)d_out;

    // Workspace layout (all offsets 16B aligned):
    float* Q  = (float*)d_ws;                    // [BL,128] f32
    float* Kp = Q  + (size_t)NBL * NU;           // [BL,128] f32
    float* Vp = Kp + (size_t)NBL * NU;           // [BL,128] f32
    _Float16* W0T = (_Float16*)(Vp + (size_t)NBL * NU);  // [64][128] f16
    _Float16* W1T = W0T + NU * NLAT;                     // [128][64] f16

    prep_weights<<<64, 128, 0, stream>>>(Wa0, Wa1, W0T, W1T);
    proj_qkv<<<NBL, NU, 0, stream>>>(x, Wq, Wk, Wv, Q, Kp, Vp);
    attn_main<<<NBL, 128, 0, stream>>>(pe, nn, Q, Kp, Vp, W0T, W1T, out);
}
